// ASLSTM_67422396612622
// MI455X (gfx1250) — matrix operand order repaired and checked by reading, not compiled, not run on hardware
//
#include <hip/hip_runtime.h>
#include <stdint.h>

// ---------------- constants ----------------
#define THRESH 0.5f
#define LENS   0.5f
#define DECAY  0.2f

#define Bsz 64
#define Tsz 512
#define Isz 256
#define Hsz 512
#define G4H 2048
#define NBLOCKS 32     // one block per h-column tile j; 4 waves = 4 batch row tiles
#define NTHREADS 128

// LDS: per-j weight set, bf16 fragments (224KB < 320KB WGP LDS)
#define SMEM_USHORTS (16384 + 3 * 32768)
#define SMEM_BYTES   (SMEM_USHORTS * 2)

typedef __attribute__((ext_vector_type(16))) __bf16          bf16x16;
typedef __attribute__((ext_vector_type(16))) unsigned short  ushort16;
typedef __attribute__((ext_vector_type(8)))  unsigned short  ushort8;
typedef __attribute__((ext_vector_type(8)))  float           floatx8;

// ---------------- small helpers ----------------
__device__ __host__ inline unsigned short f2bf(float f) {
  unsigned u = __builtin_bit_cast(unsigned, f);
  u += 0x7FFFu + ((u >> 16) & 1u);           // round-to-nearest-even
  return (unsigned short)(u >> 16);
}

__device__ inline floatx8 splat8(float v) {
  floatx8 r;
#pragma unroll
  for (int i = 0; i < 8; ++i) r[i] = v;
  return r;
}

__device__ inline float sigm(float v) { return 1.0f / (1.0f + __expf(-v)); }

// A-operand gather from a row-major bf16 [rows x ld] matrix.
// ISA 16-bit A 16x32 layout: lane l(<16) row M=l, K = hi*8 + (e<8 ? e : e+8)
// => two contiguous 16B chunks per lane.
__device__ inline bf16x16 a_gather(const unsigned short* __restrict__ src,
                                   int ld, int row0, int k0, int lane) {
  const int l = lane & 15, hi = lane >> 4;
  const unsigned short* p = src + (size_t)(row0 + l) * ld + k0 + hi * 8;
  ushort8 lo = *(const ushort8*)(p);
  ushort8 hv = *(const ushort8*)(p + 16);
  ushort16 t;
#pragma unroll
  for (int e = 0; e < 8; ++e) { t[e] = lo[e]; t[e + 8] = hv[e]; }
  return __builtin_bit_cast(bf16x16, t);
}

// Pre-packed A fragment (x): contiguous 32B per lane.
__device__ inline bf16x16 a_frag_packed(const unsigned short* __restrict__ p,
                                        int kt, int lane) {
  return __builtin_bit_cast(
      bf16x16, *(const ushort16*)(p + ((size_t)kt * 32 + lane) * 16));
}

// B fragment from LDS: layout [gate(4)][kt(KT)][lane(32)][e(16)], 32B/lane.
__device__ inline bf16x16 b_frag_lds(const unsigned short* lw, int g, int kt,
                                     int KT, int lane) {
  const ushort16* p =
      (const ushort16*)(lw + (((size_t)g * KT + kt) * 32 + lane) * 16);
  return __builtin_bit_cast(bf16x16, *p);
}

__device__ inline floatx8 wmma_bf16(bf16x16 a, bf16x16 b, floatx8 c) {
  return __builtin_amdgcn_wmma_f32_16x16x32_bf16(
      /*neg_a=*/false, a, /*neg_b=*/false, b,
      /*c_mod=*/(short)0, c, /*reuse_a=*/false, /*reuse_b=*/false);
}

// One pipeline stage: A fragment + the 4 gate B fragments.
struct Frag5 { bf16x16 a, b0, b1, b2, b3; };

// Layer-0 stage s: s<8 -> x @ Wx0 (packed A), s>=8 -> h0 @ Wh0 (gather A).
__device__ inline void load_stage_l0(Frag5& f, int s,
                                     const unsigned short* __restrict__ xpt,
                                     const unsigned short* __restrict__ h0cur,
                                     const unsigned short* lwx0,
                                     const unsigned short* lwh0,
                                     int m, int lane) {
  if (s < 8) {
    f.a  = a_frag_packed(xpt, s, lane);
    f.b0 = b_frag_lds(lwx0, 0, s, 8, lane);
    f.b1 = b_frag_lds(lwx0, 1, s, 8, lane);
    f.b2 = b_frag_lds(lwx0, 2, s, 8, lane);
    f.b3 = b_frag_lds(lwx0, 3, s, 8, lane);
  } else {
    const int kt = s - 8;
    f.a  = a_gather(h0cur, Hsz, m * 16, kt * 32, lane);
    f.b0 = b_frag_lds(lwh0, 0, kt, 16, lane);
    f.b1 = b_frag_lds(lwh0, 1, kt, 16, lane);
    f.b2 = b_frag_lds(lwh0, 2, kt, 16, lane);
    f.b3 = b_frag_lds(lwh0, 3, kt, 16, lane);
  }
}

// Layer-1 stage s: s<16 -> s0 @ Wx1, s>=16 -> h1 @ Wh1 (both gather A).
__device__ inline void load_stage_l1(Frag5& f, int s,
                                     const unsigned short* __restrict__ s0b,
                                     const unsigned short* __restrict__ h1cur,
                                     const unsigned short* lwx1,
                                     const unsigned short* lwh1,
                                     int m, int lane) {
  if (s < 16) {
    f.a  = a_gather(s0b, Hsz, m * 16, s * 32, lane);
    f.b0 = b_frag_lds(lwx1, 0, s, 16, lane);
    f.b1 = b_frag_lds(lwx1, 1, s, 16, lane);
    f.b2 = b_frag_lds(lwx1, 2, s, 16, lane);
    f.b3 = b_frag_lds(lwx1, 3, s, 16, lane);
  } else {
    const int kt = s - 16;
    f.a  = a_gather(h1cur, Hsz, m * 16, kt * 32, lane);
    f.b0 = b_frag_lds(lwh1, 0, kt, 16, lane);
    f.b1 = b_frag_lds(lwh1, 1, kt, 16, lane);
    f.b2 = b_frag_lds(lwh1, 2, kt, 16, lane);
    f.b3 = b_frag_lds(lwh1, 3, kt, 16, lane);
  }
}

// Stage one weight matrix's 4 gate slabs (for column tile j) into LDS.
// Global packed layout: [nt(128)][kt(KT)][512] ushorts, nt = gate*32 + j.
__device__ inline void stage_w(const unsigned short* __restrict__ gsrc,
                               unsigned short* ldst, int j, int KT, int tid) {
#pragma unroll
  for (int gate = 0; gate < 4; ++gate) {
    const uint4* src = (const uint4*)(gsrc + (size_t)(gate * 32 + j) * KT * 512);
    uint4* dst = (uint4*)(ldst + (size_t)gate * KT * 512);
    const int n = KT * 64;                  // 16B chunks per slab
    for (int i = tid; i < n; i += NTHREADS) dst[i] = src[i];
  }
}

// Device-wide sense-reversal barrier (bar[0]=generation, bar[1]=count).
__device__ inline void grid_sync(unsigned* bar) {
  __syncthreads();
  if (threadIdx.x == 0) {
    __threadfence();
    unsigned g = __hip_atomic_load(&bar[0], __ATOMIC_ACQUIRE, __HIP_MEMORY_SCOPE_AGENT);
    unsigned prev = __hip_atomic_fetch_add(&bar[1], 1u, __ATOMIC_ACQ_REL,
                                           __HIP_MEMORY_SCOPE_AGENT);
    if (prev == (unsigned)(NBLOCKS - 1)) {
      __hip_atomic_store(&bar[1], 0u, __ATOMIC_RELAXED, __HIP_MEMORY_SCOPE_AGENT);
      __threadfence();
      __hip_atomic_fetch_add(&bar[0], 1u, __ATOMIC_RELEASE, __HIP_MEMORY_SCOPE_AGENT);
    } else {
      while (__hip_atomic_load(&bar[0], __ATOMIC_ACQUIRE,
                               __HIP_MEMORY_SCOPE_AGENT) == g) {
        __builtin_amdgcn_s_sleep(2);
      }
    }
  }
  __syncthreads();
}

// ---------------- packing kernels (one-time, capture-safe) ----------------
// Weights fp32 [K x 2048] -> bf16 B fragments [nt(128)][kt(KT)][lane(32)][e(16)]
__global__ void pack_w_kernel(const float* __restrict__ W,
                              unsigned short* __restrict__ out, int KT) {
  size_t idx = (size_t)blockIdx.x * blockDim.x + threadIdx.x;
  size_t total = (size_t)128 * KT * 512;
  if (idx >= total) return;
  int e    = (int)(idx & 15);
  int lane = (int)((idx >> 4) & 31);
  int kt   = (int)((idx >> 9) % (size_t)KT);
  int nt   = (int)(idx / ((size_t)KT << 9));
  int col  = nt * 16 + (lane & 15);
  int k    = kt * 32 + ((lane >> 4) << 3) + ((e < 8) ? e : (e + 8));   // B layout: K = hi*8 + (e<8 ? e : e+8)
  out[idx] = f2bf(W[(size_t)k * G4H + col]);
}

// x fp32 [B][T][I] -> bf16 A fragments [t][mt(4)][kt(8)][lane(32)][e(16)]
__global__ void pack_x_kernel(const float* __restrict__ x,
                              unsigned short* __restrict__ xp) {
  size_t idx = (size_t)blockIdx.x * blockDim.x + threadIdx.x;
  if (idx >= (size_t)Tsz * 4 * 8 * 512) return;
  int e    = (int)(idx & 15);
  int lane = (int)((idx >> 4) & 31);
  int kt   = (int)((idx >> 9) & 7);
  int mt   = (int)((idx >> 12) & 3);
  int t    = (int)(idx >> 14);
  int b    = mt * 16 + (lane & 15);
  int hi   = lane >> 4;
  int koff = (e < 8) ? e : (e + 8);
  int k    = kt * 32 + hi * 8 + koff;
  xp[idx]  = f2bf(x[((size_t)b * Tsz + t) * Isz + k]);
}

// ---------------- main persistent kernel ----------------
__global__ void __launch_bounds__(NTHREADS, 1) aslstm_main(
    const float* __restrict__ bh0, const float* __restrict__ bh1,
    const unsigned short* __restrict__ xp,
    const unsigned short* __restrict__ wx0, const unsigned short* __restrict__ wh0,
    const unsigned short* __restrict__ wx1, const unsigned short* __restrict__ wh1,
    unsigned short* __restrict__ h0b,   // [2][B*H] bf16 (double-buffered)
    unsigned short* __restrict__ h1b,   // [2][B*H] bf16
    unsigned short* __restrict__ s0b,   // [B*H]    bf16
    float* __restrict__ out, unsigned* __restrict__ bar) {
  extern __shared__ unsigned short smem[];
  unsigned short* lwx0 = smem;                    // 4*8*512
  unsigned short* lwh0 = lwx0 + 4 * 8 * 512;      // 4*16*512
  unsigned short* lwx1 = lwh0 + 4 * 16 * 512;
  unsigned short* lwh1 = lwx1 + 4 * 16 * 512;

  const int tid  = threadIdx.x;
  const int lane = tid & 31;
  const int m    = tid >> 5;            // row-tile (batch), 0..3
  const int j    = blockIdx.x;          // h-column tile, 0..31
  const int l    = lane & 15, hi = lane >> 4;
  const int colbase = j * 16 + l;

  // ---- stage this block's weight columns into LDS (once) ----
  stage_w(wx0, lwx0, j, 8, tid);
  stage_w(wh0, lwh0, j, 16, tid);
  stage_w(wx1, lwx1, j, 16, tid);
  stage_w(wh1, lwh1, j, 16, tid);
  __syncthreads();

  float b0g[4], b1g[4];
#pragma unroll
  for (int g = 0; g < 4; ++g) {
    b0g[g] = bh0[g * Hsz + colbase];
    b1g[g] = bh1[g * Hsz + colbase];
  }

  // persistent per-wave state tiles (C/D accumulator layout)
  floatx8 c0 = splat8(0.f), c1 = splat8(0.f);
  floatx8 s0 = splat8(0.f), mem0 = splat8(0.f), h1r = splat8(0.f);

  for (int t = 0; t < Tsz; ++t) {
    const int par = t & 1;
    // ================= layer 0: 24-stage pipelined K loop =================
    floatx8 ai = splat8(b0g[0]), af = splat8(b0g[1]);
    floatx8 ag = splat8(b0g[2]), ao = splat8(b0g[3]);

    const unsigned short* xpt   = xp + (((size_t)t * 4 + m) * 8) * 512;
    const unsigned short* h0cur = h0b + (size_t)par * (Bsz * Hsz);

    {
      Frag5 cur, nxt;
      load_stage_l0(cur, 0, xpt, h0cur, lwx0, lwh0, m, lane);
#pragma unroll
      for (int s = 0; s < 24; ++s) {
        if (s + 1 < 24)
          load_stage_l0(nxt, s + 1, xpt, h0cur, lwx0, lwh0, m, lane);
        ai = wmma_bf16(cur.a, cur.b0, ai);
        af = wmma_bf16(cur.a, cur.b1, af);
        ag = wmma_bf16(cur.a, cur.b2, ag);
        ao = wmma_bf16(cur.a, cur.b3, ao);
        cur = nxt;
      }
    }

    unsigned short* h0nxt = h0b + (size_t)(1 - par) * (Bsz * Hsz);
#pragma unroll
    for (int e = 0; e < 8; ++e) {      // gates + ALIF neuron
      float ig = sigm(ai[e]), fg = sigm(af[e]);
      float gg = tanhf(ag[e]), og = sigm(ao[e]);
      c0[e] = fg * c0[e] + ig * gg;
      float hn  = og * tanhf(c0[e]);
      float sp  = s0[e];
      float mem = sp * DECAY * (1.0f - sp) + hn;
      float sn  = (mem > THRESH) ? 1.0f : 0.0f;
      mem0[e] = mem;
      s0[e]   = sn;
      int row = m * 16 + hi * 8 + e;
      s0b[(size_t)row * Hsz + colbase]   = f2bf(sn);
      h0nxt[(size_t)row * Hsz + colbase] = f2bf(mem);
    }
    grid_sync(bar);                    // publish s0 (and h0 for next step)

    // ================= layer 1: 32-stage pipelined K loop =================
    ai = splat8(b1g[0]); af = splat8(b1g[1]);
    ag = splat8(b1g[2]); ao = splat8(b1g[3]);

    const unsigned short* h1cur = h1b + (size_t)par * (Bsz * Hsz);
    {
      Frag5 cur, nxt;
      load_stage_l1(cur, 0, s0b, h1cur, lwx1, lwh1, m, lane);
#pragma unroll
      for (int s = 0; s < 32; ++s) {
        if (s + 1 < 32)
          load_stage_l1(nxt, s + 1, s0b, h1cur, lwx1, lwh1, m, lane);
        ai = wmma_bf16(cur.a, cur.b0, ai);
        af = wmma_bf16(cur.a, cur.b1, af);
        ag = wmma_bf16(cur.a, cur.b2, ag);
        ao = wmma_bf16(cur.a, cur.b3, ao);
        cur = nxt;
      }
    }

    unsigned short* h1nxt = h1b + (size_t)(1 - par) * (Bsz * Hsz);
#pragma unroll
    for (int e = 0; e < 8; ++e) {      // gates + output neuron (leaky integrate)
      float ig = sigm(ai[e]), fg = sigm(af[e]);
      float gg = tanhf(ag[e]), og = sigm(ao[e]);
      c1[e] = fg * c1[e] + ig * gg;
      float hc = og * tanhf(c1[e]);
      float hn = hc * DECAY + s0[e];   // xt of layer 1 is s0 (current spike)
      h1r[e] = hn;
      int row = m * 16 + hi * 8 + e;
      h1nxt[(size_t)row * Hsz + colbase]           = f2bf(hn);
      out[((size_t)row * Tsz + t) * Hsz + colbase] = hn;   // out[B,T,H]
    }
    grid_sync(bar);                    // publish h1 (and h0) for next step
  }

  // final states: out layout = out[B*T*H] | s[2,B,H] | h[2,B,H] | c[2,B,H]
  float* so = out + (size_t)Bsz * Tsz * Hsz;
  float* ho = so + 2 * Bsz * Hsz;
  float* co = ho + 2 * Bsz * Hsz;
#pragma unroll
  for (int e = 0; e < 8; ++e) {
    int row = m * 16 + hi * 8 + e;
    size_t off = (size_t)row * Hsz + colbase;
    so[off]             = s0[e];
    so[Bsz * Hsz + off] = 0.0f;        // layer-1 output neuron never spikes
    ho[off]             = mem0[e];
    ho[Bsz * Hsz + off] = h1r[e];
    co[off]             = c0[e];
    co[Bsz * Hsz + off] = c1[e];
  }
}

// ---------------- launcher ----------------
extern "C" void kernel_launch(void* const* d_in, const int* in_sizes, int n_in,
                              void* d_out, int out_size, void* d_ws,
                              size_t ws_size, hipStream_t stream) {
  const float* x   = (const float*)d_in[0];
  const float* Wx0 = (const float*)d_in[1];
  const float* Wh0 = (const float*)d_in[2];
  const float* bh0 = (const float*)d_in[3];
  const float* Wx1 = (const float*)d_in[4];
  const float* Wh1 = (const float*)d_in[5];
  const float* bh1 = (const float*)d_in[6];
  float* out = (float*)d_out;

  uint8_t* ws = (uint8_t*)d_ws;
  const size_t OFF_H0  = 256;
  const size_t OFF_H1  = OFF_H0 + (size_t)2 * Bsz * Hsz * 2;   // 131328
  const size_t OFF_S0  = OFF_H1 + (size_t)2 * Bsz * Hsz * 2;   // 262400
  const size_t OFF_XP  = OFF_S0 + (size_t)Bsz * Hsz * 2;       // 327936
  const size_t OFF_WX0 = OFF_XP + (size_t)Tsz * 4 * 8 * 512 * 2;
  const size_t OFF_WH0 = OFF_WX0 + (size_t)128 * 8  * 512 * 2;
  const size_t OFF_WX1 = OFF_WH0 + (size_t)128 * 16 * 512 * 2;
  const size_t OFF_WH1 = OFF_WX1 + (size_t)128 * 16 * 512 * 2;

  unsigned*       bar  = (unsigned*)(ws);
  unsigned short* h0b  = (unsigned short*)(ws + OFF_H0);
  unsigned short* h1b  = (unsigned short*)(ws + OFF_H1);
  unsigned short* s0b  = (unsigned short*)(ws + OFF_S0);
  unsigned short* xp   = (unsigned short*)(ws + OFF_XP);
  unsigned short* wx0p = (unsigned short*)(ws + OFF_WX0);
  unsigned short* wh0p = (unsigned short*)(ws + OFF_WH0);
  unsigned short* wx1p = (unsigned short*)(ws + OFF_WX1);
  unsigned short* wh1p = (unsigned short*)(ws + OFF_WH1);

  // zero barrier + h0/h1 state buffers (capture-safe)
  hipMemsetAsync(ws, 0, OFF_S0, stream);

  // pack x and weights into WMMA fragment layouts
  {
    size_t total = (size_t)Tsz * 4 * 8 * 512;
    pack_x_kernel<<<(unsigned)((total + 255) / 256), 256, 0, stream>>>(x, xp);
  }
  {
    size_t t8  = (size_t)128 * 8  * 512;
    size_t t16 = (size_t)128 * 16 * 512;
    pack_w_kernel<<<(unsigned)((t8  + 255) / 256), 256, 0, stream>>>(Wx0, wx0p, 8);
    pack_w_kernel<<<(unsigned)((t16 + 255) / 256), 256, 0, stream>>>(Wh0, wh0p, 16);
    pack_w_kernel<<<(unsigned)((t16 + 255) / 256), 256, 0, stream>>>(Wx1, wx1p, 16);
    pack_w_kernel<<<(unsigned)((t16 + 255) / 256), 256, 0, stream>>>(Wh1, wh1p, 16);
  }

  // allow 224KB dynamic LDS, then run the persistent cooperative kernel
  hipFuncSetAttribute((const void*)aslstm_main,
                      hipFuncAttributeMaxDynamicSharedMemorySize, SMEM_BYTES);
  aslstm_main<<<NBLOCKS, NTHREADS, SMEM_BYTES, stream>>>(
      bh0, bh1, xp, wx0p, wh0p, wx1p, wh1p, h0b, h1b, s0b, out, bar);
}
